// MultiHeadAttention_80882824118389
// MI455X (gfx1250) — matrix operand order repaired and checked by reading, not compiled, not run on hardware
//
#include <hip/hip_runtime.h>

// MultiHeadAttention forward for MI455X (gfx1250, wave32, WMMA, async-LDS).
// d_out = [ out: SEQ*BATCH*D_MODEL f32 | attn: B*H*SEQ*SEQ f32 ]
// d_ws  = [ Qp | Kp | VpT : bf16 (16MB each) | ctx bf16 16MB | wq/wk/wv/wo bf16 2MB each ] = 72MB

#define D_MODEL 1024
#define N_HEADS 16
#define D_K     64
#define SEQ     2048
#define BATCH   4

typedef __attribute__((ext_vector_type(16))) __bf16 bf16x16;
typedef __attribute__((ext_vector_type(8)))  __bf16 bf16x8;
typedef __attribute__((ext_vector_type(8)))  float  f32x8;
typedef __attribute__((ext_vector_type(4)))  float  f32x4;

// v_wmma_f32_16x16x32_bf16: D = A(16x32) * B(32x16) + C(16x16 f32)
#define WMMA_BF16(A,B,C) \
  __builtin_amdgcn_wmma_f32_16x16x32_bf16(false,(A),false,(B),(short)0,(C),false,false)

__device__ __forceinline__ f32x8 zero8(){
  f32x8 z;
  #pragma unroll
  for (int i=0;i<8;i++) z[i]=0.0f;
  return z;
}

// A-tile (16x32 bf16): lane holds row (lane%16); K = (lane/16)*16 .. +15 contiguous.
__device__ __forceinline__ bf16x16 load_a_f32(const float* rowp, int k0, int half){
  const float* p0 = rowp + k0 + half*16;
  f32x4 u0 = *(const f32x4*)(p0);
  f32x4 u1 = *(const f32x4*)(p0+4);
  f32x4 v0 = *(const f32x4*)(p0+8);
  f32x4 v1 = *(const f32x4*)(p0+12);
  bf16x16 a;
  #pragma unroll
  for (int i=0;i<4;i++){
    a[i]    = (__bf16)u0[i];  a[4+i]  = (__bf16)u1[i];
    a[8+i]  = (__bf16)v0[i];  a[12+i] = (__bf16)v1[i];
  }
  return a;
}

__device__ __forceinline__ bf16x16 load_a_bf16(const __bf16* rowp, int k0, int half){
  const __bf16* p0 = rowp + k0 + half*16;
  bf16x8 u = *(const bf16x8*)(p0);
  bf16x8 v = *(const bf16x8*)(p0+8);
  bf16x16 a;
  #pragma unroll
  for (int i=0;i<8;i++){ a[i]=u[i]; a[8+i]=v[i]; }
  return a;
}

// ---- CDNA5 async global->LDS copy (ASYNCcnt) ------------------------------
__device__ __forceinline__ void async_ld_b128(unsigned lds_off, const __bf16* g){
  asm volatile("global_load_async_to_lds_b128 %0, %1, off"
               :: "v"(lds_off), "v"(g) : "memory");
}
__device__ __forceinline__ void wait_async0(){
  asm volatile("s_wait_asynccnt 0x0" ::: "memory");
}
// Issue (no wait) async copy of a 128(n) x 32(k) bf16 weight tile into LDS [n][k].
__device__ __forceinline__ void stage_w_issue(const __bf16* W, int n0, int k0,
                                              unsigned ldsBase, int tid){
  const int rr = tid >> 2;          // 0..63
  const int cc = (tid & 3) * 8;     // 0,8,16,24 (bf16 elems; 16B chunks)
  async_ld_b128(ldsBase + (unsigned)((rr*32 + cc)*2),
                W + (size_t)(n0+rr)*D_MODEL + k0 + cc);
  async_ld_b128(ldsBase + (unsigned)(((rr+64)*32 + cc)*2),
                W + (size_t)(n0+rr+64)*D_MODEL + k0 + cc);
}

// ---------------------------------------------------------------------------
// Kernel 0: convert the four weight matrices to bf16 once.
// grid = (512, 4), block = 256; 8 elems/thread.
// ---------------------------------------------------------------------------
__global__ __launch_bounds__(256)
void convert_w_kernel(const float* __restrict__ w_q, const float* __restrict__ w_k,
                      const float* __restrict__ w_v, const float* __restrict__ w_o,
                      __bf16* __restrict__ wq, __bf16* __restrict__ wk,
                      __bf16* __restrict__ wv, __bf16* __restrict__ wo)
{
  const int which = blockIdx.y;
  const float* src = (which==0)?w_q:(which==1)?w_k:(which==2)?w_v:w_o;
  __bf16*      dst = (which==0)?wq :(which==1)?wk :(which==2)?wv :wo;
  const size_t i = ((size_t)blockIdx.x*256 + threadIdx.x)*8;
  f32x4 a = *(const f32x4*)(src+i);
  f32x4 b = *(const f32x4*)(src+i+4);
  bf16x8 o;
  #pragma unroll
  for (int k=0;k<4;k++){ o[k]=(__bf16)a[k]; o[4+k]=(__bf16)b[k]; }
  *(bf16x8*)(dst+i) = o;
}

// ---------------------------------------------------------------------------
// Kernel 1: fused Q/K/V projection. Y = X @ W^T + b.
// Q,K -> bf16 [b][h][s][dk]; V -> transposed bf16 [b][h][dk][s].
// Weight tiles double-buffered via async-to-LDS DMA overlapped with WMMA.
// grid = (64, 8, 3), block = 256 (8 waves x 16 rows).
// ---------------------------------------------------------------------------
__global__ __launch_bounds__(256)
void qkv_proj_kernel(const float* __restrict__ q_in, const float* __restrict__ k_in,
                     const float* __restrict__ v_in,
                     const __bf16* __restrict__ wq, const float* __restrict__ b_q,
                     const __bf16* __restrict__ wk, const float* __restrict__ b_k,
                     const __bf16* __restrict__ wv, const float* __restrict__ b_v,
                     __bf16* __restrict__ qp, __bf16* __restrict__ kp,
                     __bf16* __restrict__ vpT)
{
  const int which = blockIdx.z;
  const float*  X    = (which==0) ? q_in : (which==1) ? k_in : v_in;
  const __bf16* W    = (which==0) ? wq   : (which==1) ? wk   : wv;
  const float*  bias = (which==0) ? b_q  : (which==1) ? b_k  : b_v;
  __bf16* dst        = (which==0) ? qp   : (which==1) ? kp   : vpT;
  const bool transV  = (which==2);

  const int m0 = blockIdx.x * 128;        // m = b*SEQ + s
  const int n0 = blockIdx.y * 128;
  const int tid = threadIdx.x, wave = tid>>5, lane = tid&31;
  const int half = lane>>4, l16 = lane&15;

  __shared__ alignas(32) __bf16 ldsW[2][128*32];   // double buffer, 2 x 8KB
  const unsigned base0 = (unsigned)(uintptr_t)(&ldsW[0][0]);
  const unsigned base1 = (unsigned)(uintptr_t)(&ldsW[1][0]);

  const int  mrow = m0 + wave*16 + l16;
  const int  mb   = mrow / SEQ, ms = mrow % SEQ;
  const float* xrow = X + ((size_t)ms*BATCH + mb)*D_MODEL;   // input is [S,B,D]

  f32x8 acc[8];
  #pragma unroll
  for (int j=0;j<8;j++) acc[j] = zero8();

  stage_w_issue(W, n0, 0, base0, tid);               // prologue: tile 0 in flight
  int cur = 0;
  for (int k0 = 0; k0 < D_MODEL; k0 += 32) {
    wait_async0();                                   // current tile landed (this wave)
    __syncthreads();                                 // ... for all waves
    if (k0 + 32 < D_MODEL)                           // stream next tile behind compute
      stage_w_issue(W, n0, k0 + 32, cur ? base0 : base1, tid);

    bf16x16 a = load_a_f32(xrow, k0, half);
    const __bf16* wt = &ldsW[cur][0];
    #pragma unroll
    for (int j=0;j<8;j++){
      bf16x16 bm = *(const bf16x16*)&wt[(j*16 + l16)*32 + half*16];
      acc[j] = WMMA_BF16(a, bm, acc[j]);
    }
    cur ^= 1;
  }

  #pragma unroll
  for (int j=0;j<8;j++){
    const int   n  = n0 + j*16 + l16;
    const float bv = bias[n];
    const int   h  = n / D_K, dk = n % D_K;
    #pragma unroll
    for (int r=0;r<8;r++){
      const int m  = m0 + wave*16 + half*8 + r;
      const int bb = m / SEQ, ss = m % SEQ;
      const __bf16 val = (__bf16)(acc[j][r] + bv);
      if (!transV) dst[(((size_t)bb*N_HEADS + h)*SEQ + ss)*D_K + dk] = val;
      else         dst[(((size_t)bb*N_HEADS + h)*D_K + dk)*SEQ + ss] = val;
    }
  }
}

// ---------------------------------------------------------------------------
// Kernel 2: fused attention (no LDS, no barriers).
// Computes S^T = K @ Q^T per tile so scores land lane-aligned with the key
// axis: softmax stats need one shfl_xor(16); P converts in-register into the
// A-operand for P@V; attn rows store as contiguous f32x8 (32B) per lane.
// grid = (SEQ/128, H, B), block = 256 (8 waves x 16 queries).
// ---------------------------------------------------------------------------
__global__ __launch_bounds__(256)
void attention_kernel(const __bf16* __restrict__ qp, const __bf16* __restrict__ kp,
                      const __bf16* __restrict__ vpT, __bf16* __restrict__ ctx,
                      float* __restrict__ attn)
{
  const int qt = blockIdx.x, h = blockIdx.y, b = blockIdx.z;
  const int tid = threadIdx.x, wave = tid>>5, lane = tid&31;
  const int half = lane>>4, l16 = lane&15;
  const float scale = 0.125f;           // 1/sqrt(64)

  const __bf16* Qb  = qp  + (((size_t)b*N_HEADS + h)*SEQ)*D_K;
  const __bf16* Kb  = kp  + (((size_t)b*N_HEADS + h)*SEQ)*D_K;
  const __bf16* VbT = vpT + (((size_t)b*N_HEADS + h)*D_K)*SEQ;

  const int qrow = qt*128 + wave*16 + l16;     // this lane's query
  // B-operand (Q^T): col n = q (l16), K = d contiguous
  const bf16x16 qb0 = *(const bf16x16*)(Qb + (size_t)qrow*D_K + half*16);
  const bf16x16 qb1 = *(const bf16x16*)(Qb + (size_t)qrow*D_K + half*16 + 32);

  float* attnRow = attn + (((size_t)(b*N_HEADS + h)*SEQ + qrow) * SEQ);

  float mq = -1e30f, lq = 0.0f;

  // ---- pass 1: exact row max + softmax denominator ----
  for (int kt = 0; kt < SEQ; kt += 128) {
    if (kt + 128 < SEQ)
      __builtin_prefetch(Kb + (size_t)(kt + 128 + (lane<<2))*D_K, 0, 0);
    f32x8 sc[8];
    #pragma unroll
    for (int j=0;j<8;j++){
      const __bf16* krow = Kb + (size_t)(kt + j*16 + l16)*D_K;
      bf16x16 ka0 = load_a_bf16(krow, 0,  half);
      bf16x16 ka1 = load_a_bf16(krow, 32, half);
      f32x8 c = zero8();
      c = WMMA_BF16(ka0, qb0, c);   // S^T: rows = keys, col = this lane's q
      c = WMMA_BF16(ka1, qb1, c);
      sc[j] = c;
    }
    float mx = sc[0][0];
    #pragma unroll
    for (int j=0;j<8;j++)
      #pragma unroll
      for (int r=0;r<8;r++) mx = fmaxf(mx, sc[j][r]);
    mx *= scale;
    mx = fmaxf(mx, __shfl_xor(mx, 16, 32));     // combine key-halves (same q)
    const float mnew = fmaxf(mq, mx);
    float sum = 0.0f;
    #pragma unroll
    for (int j=0;j<8;j++)
      #pragma unroll
      for (int r=0;r<8;r++) sum += __expf(sc[j][r]*scale - mnew);
    sum += __shfl_xor(sum, 16, 32);
    lq = lq*__expf(mq - mnew) + sum;
    mq = mnew;
  }

  // ---- pass 2: write normalized attn once + O = P @ V ----
  f32x8 oacc[4];
  #pragma unroll
  for (int nb=0;nb<4;nb++) oacc[nb] = zero8();
  const float invl = 1.0f / lq;

  for (int kt = 0; kt < SEQ; kt += 128) {
    if (kt + 128 < SEQ){
      __builtin_prefetch(Kb  + (size_t)(kt + 128 + (lane<<2))*D_K, 0, 0);
      __builtin_prefetch(VbT + (size_t)(lane<<1)*SEQ + kt + 128,   0, 0);
    }
    f32x8 sc[8];
    #pragma unroll
    for (int j=0;j<8;j++){
      const __bf16* krow = Kb + (size_t)(kt + j*16 + l16)*D_K;
      bf16x16 ka0 = load_a_bf16(krow, 0,  half);
      bf16x16 ka1 = load_a_bf16(krow, 32, half);
      f32x8 c = zero8();
      c = WMMA_BF16(ka0, qb0, c);
      c = WMMA_BF16(ka1, qb1, c);
      sc[j] = c;
    }

    bf16x16 pa[4];     // P A-tiles: lane row = q (l16), K = keys
    #pragma unroll
    for (int j=0;j<8;j++){
      f32x8 px;
      #pragma unroll
      for (int r=0;r<8;r++) px[r] = __expf(sc[j][r]*scale - mq) * invl;
      // contiguous 8 keys for this q: single 32B store
      *(f32x8*)(attnRow + kt + j*16 + half*8) = px;
      #pragma unroll
      for (int r=0;r<8;r++) pa[j>>1][(j&1)*8 + r] = (__bf16)px[r];
    }

    #pragma unroll
    for (int nb=0;nb<4;nb++){
      f32x8 c = oacc[nb];
      #pragma unroll
      for (int t=0;t<4;t++){
        // B (V^T): col n = dim, K = keys contiguous in VpT row
        const __bf16* vp = VbT + (size_t)(nb*16 + l16)*SEQ + kt + t*32 + half*8;
        bf16x8 vlo = *(const bf16x8*)(vp);
        bf16x8 vhi = *(const bf16x8*)(vp + 16);
        bf16x16 vb;
        #pragma unroll
        for (int i=0;i<8;i++){ vb[i]=vlo[i]; vb[8+i]=vhi[i]; }
        c = WMMA_BF16(pa[t], vb, c);
      }
      oacc[nb] = c;
    }
  }

  // ctx[b][s][h*64+dk], C layout: row q = half*8 + r, col dim = l16
  #pragma unroll
  for (int nb=0;nb<4;nb++){
    #pragma unroll
    for (int r=0;r<8;r++){
      const int q = qt*128 + wave*16 + half*8 + r;
      ctx[((size_t)(b*SEQ + q))*D_MODEL + h*D_K + nb*16 + l16] = (__bf16)oacc[nb][r];
    }
  }
}

// ---------------------------------------------------------------------------
// Kernel 3: output projection. out[s][b][n] = ctx[b][s][:] @ w_o^T + b_o.
// ---------------------------------------------------------------------------
__global__ __launch_bounds__(256)
void out_proj_kernel(const __bf16* __restrict__ ctx, const __bf16* __restrict__ wo,
                     const float* __restrict__ b_o, float* __restrict__ out)
{
  const int m0 = blockIdx.x * 128;
  const int n0 = blockIdx.y * 128;
  const int tid = threadIdx.x, wave = tid>>5, lane = tid&31;
  const int half = lane>>4, l16 = lane&15;

  __shared__ alignas(32) __bf16 ldsW[2][128*32];
  const unsigned base0 = (unsigned)(uintptr_t)(&ldsW[0][0]);
  const unsigned base1 = (unsigned)(uintptr_t)(&ldsW[1][0]);

  const int mrow = m0 + wave*16 + l16;                 // m = b*SEQ + s
  const __bf16* xrow = ctx + (size_t)mrow*D_MODEL;

  f32x8 acc[8];
  #pragma unroll
  for (int j=0;j<8;j++) acc[j] = zero8();

  stage_w_issue(wo, n0, 0, base0, tid);
  int cur = 0;
  for (int k0 = 0; k0 < D_MODEL; k0 += 32) {
    wait_async0();
    __syncthreads();
    if (k0 + 32 < D_MODEL)
      stage_w_issue(wo, n0, k0 + 32, cur ? base0 : base1, tid);

    bf16x16 a = load_a_bf16(xrow, k0, half);
    const __bf16* wt = &ldsW[cur][0];
    #pragma unroll
    for (int j=0;j<8;j++){
      bf16x16 bm = *(const bf16x16*)&wt[(j*16 + l16)*32 + half*16];
      acc[j] = WMMA_BF16(a, bm, acc[j]);
    }
    cur ^= 1;
  }

  #pragma unroll
  for (int j=0;j<8;j++){
    const int   n  = n0 + j*16 + l16;
    const float bv = b_o[n];
    #pragma unroll
    for (int r=0;r<8;r++){
      const int m  = m0 + wave*16 + half*8 + r;
      const int bb = m / SEQ, ss = m % SEQ;
      out[((size_t)ss*BATCH + bb)*D_MODEL + n] = acc[j][r] + bv;   // [S,B,D]
    }
  }
}

// ---------------------------------------------------------------------------
extern "C" void kernel_launch(void* const* d_in, const int* in_sizes, int n_in,
                              void* d_out, int out_size, void* d_ws, size_t ws_size,
                              hipStream_t stream)
{
  (void)in_sizes; (void)n_in; (void)out_size; (void)ws_size;

  const float* query = (const float*)d_in[0];
  const float* key_  = (const float*)d_in[1];
  const float* value = (const float*)d_in[2];
  const float* w_q = (const float*)d_in[3];
  const float* b_q = (const float*)d_in[4];
  const float* w_k = (const float*)d_in[5];
  const float* b_k = (const float*)d_in[6];
  const float* w_v = (const float*)d_in[7];
  const float* b_v = (const float*)d_in[8];
  const float* w_o = (const float*)d_in[9];
  const float* b_o = (const float*)d_in[10];

  float* out  = (float*)d_out;
  float* attn = out + (size_t)SEQ*BATCH*D_MODEL;

  const size_t NE = (size_t)BATCH*N_HEADS*SEQ*D_K;   // 8,388,608
  const size_t NW = (size_t)D_MODEL*D_MODEL;         // 1,048,576
  __bf16* qp  = (__bf16*)d_ws;
  __bf16* kp  = qp  + NE;
  __bf16* vpT = kp  + NE;
  __bf16* ctx = vpT + NE;
  __bf16* wq  = ctx + NE;
  __bf16* wk  = wq  + NW;
  __bf16* wv  = wk  + NW;
  __bf16* wo  = wv  + NW;                            // total ws: 72MB

  convert_w_kernel<<<dim3(512, 4), 256, 0, stream>>>(
      w_q, w_k, w_v, w_o, wq, wk, wv, wo);
  qkv_proj_kernel<<<dim3(64, 8, 3), 256, 0, stream>>>(
      query, key_, value, wq, b_q, wk, b_k, wv, b_v, qp, kp, vpT);
  attention_kernel<<<dim3(SEQ/128, N_HEADS, BATCH), 256, 0, stream>>>(
      qp, kp, vpT, ctx, attn);
  out_proj_kernel<<<dim3(64, 8, 1), 256, 0, stream>>>(ctx, wo, b_o, out);
}
